// GraphCNN_352187319087
// MI455X (gfx1250) — matrix operand order repaired and checked by reading, not compiled, not run on hardware
//
#include <hip/hip_runtime.h>

typedef __attribute__((ext_vector_type(16))) __bf16 v16bf;
typedef __attribute__((ext_vector_type(8)))  __bf16 v8bf;
typedef __attribute__((ext_vector_type(8)))  float  v8f;

#define N_NODES 90
#define NP      96      // padded node dim (3 k-tiles of 32)
#define HID     128
#define ADJ_LD  104     // bf16 elements; 208B rows -> conflict-free, 16B aligned
#define H_LD    136     // 272B rows
#define T_LD    104

// LDS layout (in __bf16 elements)
#define OFF_ADJ 0
#define OFF_H0  (OFF_ADJ + NP * ADJ_LD)   //  9984
#define OFF_H0T (OFF_H0  + NP * H_LD)     // 23040
#define OFF_R1  (OFF_H0T + HID * T_LD)    // 36352
#define OFF_U   (OFF_R1  + NP * H_LD)     // 49408 (T, then U)
#define OFF_XT  (OFF_U   + NP * H_LD)     // 62464
#define SMEM_ELEMS (OFF_XT + HID * T_LD)  // 75776
#define SMEM_BYTES (SMEM_ELEMS * 2)       // 151552 B < 160KB (2 WGs/WGP fit in 320KB)

union AB16 { v16bf v; v8bf h[2]; };

// C(96 x 16) column-block GEMM: this wave computes N-tile `nt`.
// A: row-major [96 x K] in LDS (lda elems). Bt: row-major transposed B,
// Bt[N][Kpad] (ldb elems) — B[k][n] = Bt[n][k] — in LDS or global (L2-hot).
// Accumulates into acc[6] (M-tiles 0..5).
__device__ __forceinline__ void gemm6(const __bf16* __restrict__ A, int lda,
                                      const __bf16* __restrict__ Bt, int ldb,
                                      int ktiles, int nt, int lane, v8f acc[6])
{
    const int n  = lane & 15;
    const int hi = lane >> 4;
    for (int kt = 0; kt < ktiles; ++kt) {
        // B fragment: lane holds column n, K = kt*32 + {8hi..8hi+7, 16+8hi..16+8hi+7}
        AB16 b;
        const __bf16* pb = Bt + (size_t)(nt * 16 + n) * ldb + kt * 32 + hi * 8;
        b.h[0] = *(const v8bf*)pb;
        b.h[1] = *(const v8bf*)(pb + 16);
        const __bf16* pa0 = A + (size_t)n * lda + kt * 32 + hi * 8;
#pragma unroll
        for (int mt = 0; mt < 6; ++mt) {
            // A fragment: lane holds row mt*16+n, K = kt*32 + {8hi..8hi+7, 16+8hi..16+8hi+7}
            AB16 a;
            const __bf16* pa = pa0 + (size_t)mt * 16 * lda;
            a.h[0] = *(const v8bf*)pa;
            a.h[1] = *(const v8bf*)(pa + 16);
            acc[mt] = __builtin_amdgcn_wmma_f32_16x16x32_bf16(
                false, a.v, false, b.v, (short)0, acc[mt], false, false);
        }
    }
}

__device__ __forceinline__ void acc_fill(v8f acc[6], float bv)
{
#pragma unroll
    for (int mt = 0; mt < 6; ++mt)
#pragma unroll
        for (int r = 0; r < 8; ++r) acc[mt][r] = bv;
}

// Convert W[K][N] f32 -> Wt[N][Kpad] bf16 (zero-padded K)
__global__ void cvt_weight(const float* __restrict__ src, __bf16* __restrict__ dst,
                           int K, int N, int Kpad)
{
    int idx = blockIdx.x * 256 + threadIdx.x;
    int total = N * Kpad;
    if (idx >= total) return;
    int nrow = idx / Kpad;
    int k = idx - nrow * Kpad;
    float v = (k < K) ? src[(size_t)k * N + nrow] : 0.0f;
    dst[idx] = (__bf16)v;
}

__global__ __launch_bounds__(256)
void graphcnn_main(const float* __restrict__ adj,
                   const __bf16* __restrict__ w0t,   // [128][96]
                   const __bf16* __restrict__ wr1t,  // [128][128]
                   const __bf16* __restrict__ w1t,   // [128][128]
                   const __bf16* __restrict__ wr2t,  // [128][128]
                   const __bf16* __restrict__ w2t,   // [128][128]
                   const float* __restrict__ b0,
                   const float* __restrict__ br1,
                   const float* __restrict__ b1,
                   const float* __restrict__ b2,
                   const float* __restrict__ br2,
                   float* __restrict__ out)
{
    extern __shared__ __bf16 smem[];
    __bf16* sAdj = smem + OFF_ADJ;   // [96][104]  bf16 adj, zero-padded
    __bf16* sH0  = smem + OFF_H0;    // [96][136]  H0 row-major (A use); later Y
    __bf16* sH0t = smem + OFF_H0T;   // [128][104] H0 transposed (B use)
    __bf16* sR1  = smem + OFF_R1;    // [96][136]
    __bf16* sU   = smem + OFF_U;     // [96][136]  T, then U
    __bf16* sXt  = smem + OFF_XT;    // [128][104] X transposed (B use)

    const int b    = blockIdx.x;
    const int tid  = threadIdx.x;
    const int wave = tid >> 5;
    const int lane = tid & 31;
    const int n    = lane & 15;
    const int hi   = lane >> 4;
    const int col  = wave * 16 + n;  // N index for N=128 stages

    // ---- load adj -> LDS bf16, zero-pad to 96x96 ----
    const float* ab = adj + (size_t)b * (N_NODES * N_NODES);
    for (int idx = tid; idx < NP * NP; idx += 256) {
        int i = idx / NP, j = idx - i * NP;
        float v = (i < N_NODES && j < N_NODES) ? ab[i * N_NODES + j] : 0.0f;
        sAdj[i * ADJ_LD + j] = (__bf16)v;
    }
    __syncthreads();

    v8f acc[6];

    // ---- Stage 1: H0 = adj @ W0 + b0 (K=96) -> sH0 (row-major) + sH0t ----
    acc_fill(acc, b0[col]);
    gemm6(sAdj, ADJ_LD, w0t, NP, 3, wave, lane, acc);
#pragma unroll
    for (int mt = 0; mt < 6; ++mt) {
        v8bf pk;
#pragma unroll
        for (int r = 0; r < 8; ++r) {
            int m = mt * 16 + hi * 8 + r;
            __bf16 hv = (__bf16)acc[mt][r];
            sH0[m * H_LD + col] = hv;
            pk[r] = hv;
        }
        *(v8bf*)(sH0t + (size_t)col * T_LD + mt * 16 + hi * 8) = pk;
    }
    __syncthreads();

    // ---- Stage 2: R1 = H0 @ Wr1 + br1 (K=128) -> sR1 ----
    acc_fill(acc, br1[col]);
    gemm6(sH0, H_LD, wr1t, HID, 4, wave, lane, acc);
#pragma unroll
    for (int mt = 0; mt < 6; ++mt)
#pragma unroll
        for (int r = 0; r < 8; ++r)
            sR1[(mt * 16 + hi * 8 + r) * H_LD + col] = (__bf16)acc[mt][r];

    // ---- Stage 3: T = adj @ H0 (B from sH0t, K=96) -> sU (no barrier needed) ----
    acc_fill(acc, 0.0f);
    gemm6(sAdj, ADJ_LD, sH0t, T_LD, 3, wave, lane, acc);
#pragma unroll
    for (int mt = 0; mt < 6; ++mt)
#pragma unroll
        for (int r = 0; r < 8; ++r)
            sU[(mt * 16 + hi * 8 + r) * H_LD + col] = (__bf16)acc[mt][r];
    __syncthreads();

    // ---- Stage 4: X = relu(T @ W1 + b1) + R1  -> sXt (transposed) ----
    acc_fill(acc, b1[col]);
    gemm6(sU, H_LD, w1t, HID, 4, wave, lane, acc);
#pragma unroll
    for (int mt = 0; mt < 6; ++mt) {
        v8bf pk;
#pragma unroll
        for (int r = 0; r < 8; ++r) {
            int m = mt * 16 + hi * 8 + r;
            float rv = (float)sR1[m * H_LD + col];
            pk[r] = (__bf16)(fmaxf(acc[mt][r], 0.0f) + rv);
        }
        *(v8bf*)(sXt + (size_t)col * T_LD + mt * 16 + hi * 8) = pk;
    }
    __syncthreads();

    // ---- Stage 5: U = adj @ X (B from sXt, K=96) -> sU (overwrite T) ----
    acc_fill(acc, 0.0f);
    gemm6(sAdj, ADJ_LD, sXt, T_LD, 3, wave, lane, acc);
#pragma unroll
    for (int mt = 0; mt < 6; ++mt)
#pragma unroll
        for (int r = 0; r < 8; ++r)
            sU[(mt * 16 + hi * 8 + r) * H_LD + col] = (__bf16)acc[mt][r];
    __syncthreads();

    // ---- Stage 6: Y = relu(U @ W2 + R1 @ Wr2 + b2 + br2) -> sH0 (reuse) ----
    acc_fill(acc, b2[col] + br2[col]);
    gemm6(sU,  H_LD, w2t,  HID, 4, wave, lane, acc);
    gemm6(sR1, H_LD, wr2t, HID, 4, wave, lane, acc);
#pragma unroll
    for (int mt = 0; mt < 6; ++mt)
#pragma unroll
        for (int r = 0; r < 8; ++r)
            sH0[(mt * 16 + hi * 8 + r) * H_LD + col] = (__bf16)fmaxf(acc[mt][r], 0.0f);
    __syncthreads();

    // ---- Stage 7: Z = Y @ Y^T (K=128); waves 0..5 own N-tiles 0..5 ----
    if (wave < 6) {
        acc_fill(acc, 0.0f);
        // B[k][j] = Y[j][k]  =>  "Bt" is row-major Y itself
        gemm6(sH0, H_LD, sH0, H_LD, 4, wave, lane, acc);
        float* ob = out + (size_t)b * (N_NODES * N_NODES);
        int j = wave * 16 + n;
        if (j < N_NODES) {
#pragma unroll
            for (int mt = 0; mt < 6; ++mt)
#pragma unroll
                for (int r = 0; r < 8; ++r) {
                    int i = mt * 16 + hi * 8 + r;
                    if (i < N_NODES) ob[i * N_NODES + j] = acc[mt][r];
                }
        }
    }
}

extern "C" void kernel_launch(void* const* d_in, const int* in_sizes, int n_in,
                              void* d_out, int out_size, void* d_ws, size_t ws_size,
                              hipStream_t stream)
{
    const float* adj = (const float*)d_in[0];
    const float* W0  = (const float*)d_in[1];
    const float* b0  = (const float*)d_in[2];
    const float* W1  = (const float*)d_in[3];
    const float* b1  = (const float*)d_in[4];
    const float* W2  = (const float*)d_in[5];
    const float* b2  = (const float*)d_in[6];
    const float* Wr1 = (const float*)d_in[7];
    const float* br1 = (const float*)d_in[8];
    const float* Wr2 = (const float*)d_in[9];
    const float* br2 = (const float*)d_in[10];
    float* out = (float*)d_out;

    const int B = in_sizes[0] / (N_NODES * N_NODES);

    // workspace: bf16 transposed padded weights (L2-resident, ~152KB)
    __bf16* ws   = (__bf16*)d_ws;
    __bf16* w0t  = ws;                    // 128 x 96
    __bf16* wr1t = w0t  + 128 * NP;       // 128 x 128
    __bf16* w1t  = wr1t + 128 * 128;
    __bf16* wr2t = w1t  + 128 * 128;
    __bf16* w2t  = wr2t + 128 * 128;

    cvt_weight<<<(128 * NP  + 255) / 256, 256, 0, stream>>>(W0,  w0t,  N_NODES, HID, NP);
    cvt_weight<<<(128 * 128 + 255) / 256, 256, 0, stream>>>(Wr1, wr1t, HID, HID, HID);
    cvt_weight<<<(128 * 128 + 255) / 256, 256, 0, stream>>>(W1,  w1t,  HID, HID, HID);
    cvt_weight<<<(128 * 128 + 255) / 256, 256, 0, stream>>>(Wr2, wr2t, HID, HID, HID);
    cvt_weight<<<(128 * 128 + 255) / 256, 256, 0, stream>>>(W2,  w2t,  HID, HID, HID);

    graphcnn_main<<<B, 256, SMEM_BYTES, stream>>>(adj, w0t, wr1t, w1t, wr2t, w2t,
                                                  b0, br1, b1, b2, br2, out);
}
